// MultiHeadAttentionWithRelativePosition_21242908246423
// MI455X (gfx1250) — matrix operand order repaired and checked by reading, not compiled, not run on hardware
//
#include <hip/hip_runtime.h>
#include <hip/hip_bf16.h>

typedef __attribute__((ext_vector_type(16))) __bf16 v16bf;
typedef __attribute__((ext_vector_type(8)))  float  v8f;

#define BN 4
#define SN 512
#define DN 512
#define HN 8
#define DKN 64
#define PN 16

// ---------------- WMMA fragment helpers (wave32) ----------------
// A 16x32 bf16: lane (g = l>>4, r = l&15) holds row r; elems 0-7 -> K=k0+8g+0..7,
// elems 8-15 -> K=k0+16+8g+0..7  (ISA 7.12.2 16-bit A table)
__device__ __forceinline__ v16bf load_a_frag(const __bf16* m, int ld, int row0, int k0) {
  int l = threadIdx.x & 31;
  const __bf16* p = m + (size_t)(row0 + (l & 15)) * ld + k0 + ((l >> 4) << 3);
  v16bf f;
  *(uint4*)&f       = *(const uint4*)p;        // 8 bf16
  *((uint4*)&f + 1) = *(const uint4*)(p + 16); // 8 bf16
  return f;
}

// B 32x16 bf16 with B[k][n] = Src[n][k] (Src row-major over k): lane n holds col n;
// elems 0-7 -> K = k0 + 8g + 0..7, elems 8-15 -> K = k0 + 16 + 8g + 0..7 (same as A)
__device__ __forceinline__ v16bf load_b_frag(const __bf16* m, int ld, int row0, int k0) {
  int l = threadIdx.x & 31;
  const __bf16* p = m + (size_t)(row0 + (l & 15)) * ld + k0 + ((l >> 4) << 3);
  v16bf f;
  *(uint4*)&f       = *(const uint4*)p;
  *((uint4*)&f + 1) = *(const uint4*)(p + 16);
  return f;
}

__device__ __forceinline__ v8f wmma_bf16(v16bf a, v16bf b, v8f c) {
  return __builtin_amdgcn_wmma_f32_16x16x32_bf16(false, a, false, b, (short)0, c, false, false);
}

// ---------------- elementwise f32 -> bf16 ----------------
__global__ void cvt_bf16_kernel(const float* __restrict__ src, __bf16* __restrict__ dst, int n) {
  int i = blockIdx.x * blockDim.x + threadIdx.x;
  if (i < n) dst[i] = (__bf16)src[i];
}

// ---------------- rel_table.mean(-1) -> tmean[33] ----------------
__global__ void table_mean_kernel(const float* __restrict__ tab, float* __restrict__ tm) {
  int t = threadIdx.x;
  if (t < 2 * PN + 1) {
    float s = 0.f;
    for (int e = 0; e < DKN; ++e) s += tab[t * DKN + e];
    tm[t] = s * (1.f / (float)DKN);
  }
}

// ---------------- generic projection GEMM: Y = X @ W^T + bias ----------------
// X: [2048 x 512] bf16 row-major, W: [512 x 512] bf16 row-major (row n = out-feature n)
// mode 0: bf16 out to [b,h,s,dk]   (Q, K)
// mode 1: bf16 out to [b,h,dk,s]   (V transposed)
// mode 2: f32  out = acc + bias + resid, token-major [m, n]  (output projection)
__global__ void proj_gemm_kernel(const __bf16* __restrict__ X, const __bf16* __restrict__ W,
                                 const float* __restrict__ bias, const float* __restrict__ resid,
                                 __bf16* __restrict__ out_bf, float* __restrict__ out_f, int mode) {
  int id = blockIdx.x;
  int nt = id & 31, it = id >> 5;
  v8f c = {};
  for (int k0 = 0; k0 < DN; k0 += 32) {
    v16bf a = load_a_frag(X, DN, it * 16, k0);
    v16bf b = load_b_frag(W, DN, nt * 16, k0);
    c = wmma_bf16(a, b, c);
  }
  int lane = threadIdx.x & 31;
  int n  = nt * 16 + (lane & 15);
  int mb = it * 16 + ((lane >> 4) << 3);
  float bv = bias[n];
#pragma unroll
  for (int v = 0; v < 8; ++v) {
    int m = mb + v;  // token 0..2047
    float val = c[v] + bv;
    if (mode == 2) {
      out_f[(size_t)m * DN + n] = val + resid[(size_t)m * DN + n];
    } else {
      int b_ = m >> 9, s = m & 511, h = n >> 6, dk = n & 63;
      if (mode == 0)
        out_bf[(((size_t)(b_ * HN + h)) * SN + s) * DKN + dk] = (__bf16)val;
      else
        out_bf[(((size_t)(b_ * HN + h)) * DKN + dk) * SN + s] = (__bf16)val;
    }
  }
}

// ---------------- scores = Q K^T / sqrt(dk) ----------------
__global__ void scores_qk_kernel(const __bf16* __restrict__ Q, const __bf16* __restrict__ K,
                                 float* __restrict__ scores) {
  int id = blockIdx.x;
  int jt = id & 31, it = (id >> 5) & 31, h = (id >> 10) & 7, b = id >> 13;
  const __bf16* Qb = Q + ((size_t)(b * HN + h) * SN) * DKN;
  const __bf16* Kb = K + ((size_t)(b * HN + h) * SN) * DKN;
  v8f c = {};
  for (int k0 = 0; k0 < DKN; k0 += 32) {
    v16bf a  = load_a_frag(Qb, DKN, it * 16, k0);
    v16bf bb = load_b_frag(Kb, DKN, jt * 16, k0);
    c = wmma_bf16(a, bb, c);
  }
  int lane = threadIdx.x & 31;
  int n  = jt * 16 + (lane & 15);
  int mb = it * 16 + ((lane >> 4) << 3);
  float* srow = scores + ((size_t)(b * HN + h) * SN) * SN;
#pragma unroll
  for (int v = 0; v < 8; ++v)
    srow[(size_t)(mb + v) * SN + n] = c[v] * 0.125f;  // 1/sqrt(64)
}

// ---------------- relative-position bias: scores[b,h,i,j] += sum_d Q[b,h,i,d]*tmean[idx[b,i,j,d]]
// One wave per (b, i, j-tile). A rows = heads (8 used of 16), B = gathered table values.
__global__ void bias_add_kernel(const __bf16* __restrict__ Q, const int* __restrict__ rp,
                                const float* __restrict__ tm, float* __restrict__ scores) {
  __shared__ float lt[2 * PN + 1];
  for (int t = threadIdx.x; t < 2 * PN + 1; t += 32) lt[t] = tm[t];
  __syncthreads();

  int id = blockIdx.x;
  int jt = id & 31, i = (id >> 5) & 511, b = id >> 14;
  int lane = threadIdx.x & 31;
  int r = lane & 15, g = lane >> 4;

  v8f c = {};
  for (int k0 = 0; k0 < DKN; k0 += 32) {
    v16bf a;
    {
      int hrow = (r < 8) ? r : 0;
      const __bf16* p = Q + (((size_t)(b * HN + hrow)) * SN + i) * DKN + k0 + (g << 3);
#pragma unroll
      for (int e = 0; e < 8; ++e) {
        float lo = (r < 8) ? (float)p[e]      : 0.f;
        float hi = (r < 8) ? (float)p[16 + e] : 0.f;
        a[e]     = (__bf16)lo;
        a[8 + e] = (__bf16)hi;
      }
    }
    v16bf bb;
    {
      int j = jt * 16 + r;
      const int* ip = rp + (((size_t)(b * SN + i)) * SN + j) * DKN + k0 + (g << 3);
#pragma unroll
      for (int e = 0; e < 16; ++e) {
        int ix = ip[e < 8 ? e : e + 8];
        ix = (ix < -PN) ? -PN : (ix > PN ? PN : ix);
        bb[e] = (__bf16)lt[ix + PN];
      }
    }
    c = wmma_bf16(a, bb, c);
  }
  if (g == 0) {  // rows 0..7 = heads
    int j = jt * 16 + r;
#pragma unroll
    for (int v = 0; v < 8; ++v) {
      size_t off = (((size_t)(b * HN + v)) * SN + i) * SN + j;
      scores[off] += c[v];
    }
  }
}

// ---------------- softmax over j (wave per row), mask, dual write f32 + bf16 ----------------
__global__ void softmax_kernel(float* __restrict__ scores, const int* __restrict__ mask,
                               __bf16* __restrict__ attn_bf) {
  int w = threadIdx.x >> 5, lane = threadIdx.x & 31;
  int row = blockIdx.x * 8 + w;      // 0 .. B*H*S-1
  int b = row >> 12;                 // / (H*S)
  float* s = scores + (size_t)row * SN;
  const int* mrow = mask + b * SN;

  float vals[16];
  float vmax = -3.4e38f;
#pragma unroll
  for (int t = 0; t < 16; ++t) {
    int j = lane + (t << 5);
    float x = s[j];
    if (mrow[j] == 0) x = -1e9f;
    vals[t] = x;
    vmax = fmaxf(vmax, x);
  }
  for (int o = 16; o > 0; o >>= 1) vmax = fmaxf(vmax, __shfl_xor(vmax, o, 32));
  float sum = 0.f;
#pragma unroll
  for (int t = 0; t < 16; ++t) { vals[t] = __expf(vals[t] - vmax); sum += vals[t]; }
  for (int o = 16; o > 0; o >>= 1) sum += __shfl_xor(sum, o, 32);
  float inv = 1.f / sum;
#pragma unroll
  for (int t = 0; t < 16; ++t) {
    int j = lane + (t << 5);
    float p = vals[t] * inv;
    s[j] = p;
    attn_bf[(size_t)row * SN + j] = (__bf16)p;
  }
}

// ---------------- ctx = attn @ V (V pre-transposed [b,h,dk,s]) ----------------
__global__ void ctx_gemm_kernel(const __bf16* __restrict__ attn_bf, const __bf16* __restrict__ Vt,
                                __bf16* __restrict__ ctx_bf) {
  int id = blockIdx.x;
  int nt = id & 3, it = (id >> 2) & 31, h = (id >> 7) & 7, b = id >> 10;
  const __bf16* A  = attn_bf + ((size_t)(b * HN + h) * SN) * SN;
  const __bf16* Vb = Vt      + ((size_t)(b * HN + h) * DKN) * SN;
  v8f c = {};
  for (int k0 = 0; k0 < SN; k0 += 32) {
    v16bf a  = load_a_frag(A,  SN, it * 16, k0);
    v16bf bb = load_b_frag(Vb, SN, nt * 16, k0);
    c = wmma_bf16(a, bb, c);
  }
  int lane = threadIdx.x & 31;
  int n   = nt * 16 + (lane & 15);          // dk within head
  int mb  = it * 16 + ((lane >> 4) << 3);   // token s within b
  int col = h * DKN + n;
#pragma unroll
  for (int v = 0; v < 8; ++v)
    ctx_bf[((size_t)(b * SN) + mb + v) * DN + col] = (__bf16)c[v];
}

// ---------------- LayerNorm (wave per token row) ----------------
__global__ void layernorm_kernel(const float* __restrict__ x, const float* __restrict__ gamma,
                                 const float* __restrict__ beta, float* __restrict__ y) {
  int w = threadIdx.x >> 5, lane = threadIdx.x & 31;
  int row = blockIdx.x * 8 + w;  // 0..2047
  const float* xr = x + (size_t)row * DN;
  float v[16];
  float s = 0.f, ss = 0.f;
#pragma unroll
  for (int t = 0; t < 16; ++t) {
    float xx = xr[lane + (t << 5)];
    v[t] = xx; s += xx; ss += xx * xx;
  }
  for (int o = 16; o > 0; o >>= 1) { s += __shfl_xor(s, o, 32); ss += __shfl_xor(ss, o, 32); }
  float mu   = s * (1.f / (float)DN);
  float var  = ss * (1.f / (float)DN) - mu * mu;
  float rstd = rsqrtf(var + 1e-5f);
#pragma unroll
  for (int t = 0; t < 16; ++t) {
    int j = lane + (t << 5);
    y[(size_t)row * DN + j] = (v[t] - mu) * rstd * gamma[j] + beta[j];
  }
}

extern "C" void kernel_launch(void* const* d_in, const int* in_sizes, int n_in,
                              void* d_out, int out_size, void* d_ws, size_t ws_size,
                              hipStream_t stream) {
  (void)in_sizes; (void)n_in; (void)out_size; (void)ws_size;
  const float* query = (const float*)d_in[0];
  const float* key   = (const float*)d_in[1];
  const float* value = (const float*)d_in[2];
  const float* Wq = (const float*)d_in[3];
  const float* bq = (const float*)d_in[4];
  const float* Wk = (const float*)d_in[5];
  const float* bk = (const float*)d_in[6];
  const float* Wv = (const float*)d_in[7];
  const float* bv = (const float*)d_in[8];
  const float* Wo = (const float*)d_in[9];
  const float* bo = (const float*)d_in[10];
  const float* rel_table = (const float*)d_in[11];
  const float* gamma = (const float*)d_in[12];
  const float* beta  = (const float*)d_in[13];
  const int* mask   = (const int*)d_in[14];
  const int* relpos = (const int*)d_in[15];

  float* y_out    = (float*)d_out;                    // [B,S,D]
  float* attn_out = y_out + (size_t)BN * SN * DN;     // [B,H,S,S]

  char* ws = (char*)d_ws;
  size_t off = 0;
  auto alloc = [&](size_t bytes) -> void* {
    void* p = ws + off;
    off += bytes;
    off = (off + 255) & ~(size_t)255;
    return p;
  };
  const size_t nAct = (size_t)BN * SN * DN;   // 1,048,576
  const size_t nW   = (size_t)DN * DN;        // 262,144
  __bf16* xq_bf = (__bf16*)alloc(nAct * 2);
  __bf16* xk_bf = (__bf16*)alloc(nAct * 2);
  __bf16* xv_bf = (__bf16*)alloc(nAct * 2);
  __bf16* wq_bf = (__bf16*)alloc(nW * 2);
  __bf16* wk_bf = (__bf16*)alloc(nW * 2);
  __bf16* wv_bf = (__bf16*)alloc(nW * 2);
  __bf16* wo_bf = (__bf16*)alloc(nW * 2);
  __bf16* Qbf   = (__bf16*)alloc(nAct * 2);
  __bf16* Kbf   = (__bf16*)alloc(nAct * 2);
  __bf16* Vtbf  = (__bf16*)alloc(nAct * 2);
  float*  tmean = (float*)alloc((2 * PN + 1) * sizeof(float));
  __bf16* attn_bf = (__bf16*)alloc((size_t)BN * HN * SN * SN * 2);  // 16 MB
  __bf16* ctx_bf  = (__bf16*)alloc(nAct * 2);
  float*  xres    = (float*)alloc(nAct * 4);

  // 1) convert to bf16
  cvt_bf16_kernel<<<(int)((nAct + 255) / 256), 256, 0, stream>>>(query, xq_bf, (int)nAct);
  cvt_bf16_kernel<<<(int)((nAct + 255) / 256), 256, 0, stream>>>(key,   xk_bf, (int)nAct);
  cvt_bf16_kernel<<<(int)((nAct + 255) / 256), 256, 0, stream>>>(value, xv_bf, (int)nAct);
  cvt_bf16_kernel<<<(int)((nW + 255) / 256), 256, 0, stream>>>(Wq, wq_bf, (int)nW);
  cvt_bf16_kernel<<<(int)((nW + 255) / 256), 256, 0, stream>>>(Wk, wk_bf, (int)nW);
  cvt_bf16_kernel<<<(int)((nW + 255) / 256), 256, 0, stream>>>(Wv, wv_bf, (int)nW);
  cvt_bf16_kernel<<<(int)((nW + 255) / 256), 256, 0, stream>>>(Wo, wo_bf, (int)nW);
  table_mean_kernel<<<1, 64, 0, stream>>>(rel_table, tmean);

  // 2) QKV projections (WMMA)
  proj_gemm_kernel<<<4096, 32, 0, stream>>>(xq_bf, wq_bf, bq, nullptr, Qbf,  nullptr, 0);
  proj_gemm_kernel<<<4096, 32, 0, stream>>>(xk_bf, wk_bf, bk, nullptr, Kbf,  nullptr, 0);
  proj_gemm_kernel<<<4096, 32, 0, stream>>>(xv_bf, wv_bf, bv, nullptr, Vtbf, nullptr, 1);

  // 3) scores = QK^T/8, then add relative-position bias (WMMA over heads)
  scores_qk_kernel<<<BN * HN * 32 * 32, 32, 0, stream>>>(Qbf, Kbf, attn_out);
  bias_add_kernel<<<BN * SN * 32, 32, 0, stream>>>(Qbf, relpos, tmean, attn_out);

  // 4) masked softmax -> attn (f32 in d_out) + bf16 copy
  softmax_kernel<<<BN * HN * SN / 8, 256, 0, stream>>>(attn_out, mask, attn_bf);

  // 5) ctx = attn @ V (WMMA), output projection + residual, LayerNorm
  ctx_gemm_kernel<<<BN * HN * 32 * 4, 32, 0, stream>>>(attn_bf, Vtbf, ctx_bf);
  proj_gemm_kernel<<<4096, 32, 0, stream>>>(ctx_bf, wo_bf, bo, query, nullptr, xres, 2);
  layernorm_kernel<<<BN * SN / 8, 256, 0, stream>>>(xres, gamma, beta, y_out);
}
